// SoftHungarian_27831388078328
// MI455X (gfx1250) — matrix operand order repaired and checked by reading, not compiled, not run on hardware
//
#include <hip/hip_runtime.h>
#include <hip/hip_bf16.h>

typedef __attribute__((ext_vector_type(16))) _Float16 v16h;
typedef __attribute__((ext_vector_type(8)))  _Float16 v8h;
typedef __attribute__((ext_vector_type(8)))  float    v8f;

#define NB  8      // batches
#define NN  384    // nodes per graph
#define DD  1024   // feature dim
#define NROW (2*NB*NN)   // 6144 stacked rows (t1 then t2)
#define NEG_SLOPE 0.2f

// ---------------------------------------------------------------------------
// fp32 -> fp16 elementwise convert
// ---------------------------------------------------------------------------
__global__ __launch_bounds__(256) void cvt_f16_kernel(const float* __restrict__ in,
                                                      _Float16* __restrict__ out,
                                                      long long n) {
    long long i = (long long)blockIdx.x * blockDim.x + threadIdx.x;
    if (i < n) out[i] = (_Float16)in[i];
}

// ---------------------------------------------------------------------------
// Row-normalize (L2 over D) fp32 row -> fp16 row. One block per row.
// ---------------------------------------------------------------------------
__global__ __launch_bounds__(256) void rownorm_f16_kernel(const float* __restrict__ in,
                                                          _Float16* __restrict__ out) {
    __shared__ float red[256];
    const long long row = blockIdx.x;
    const float* r = in + row * (long long)DD;
    float s = 0.f;
    for (int c = threadIdx.x; c < DD; c += 256) { float x = r[c]; s += x * x; }
    red[threadIdx.x] = s; __syncthreads();
    for (int w = 128; w > 0; w >>= 1) {
        if (threadIdx.x < w) red[threadIdx.x] += red[threadIdx.x + w];
        __syncthreads();
    }
    const float inv = 1.0f / fmaxf(sqrtf(red[0]), 1e-12f);
    _Float16* o = out + row * (long long)DD;
    for (int c = threadIdx.x; c < DD; c += 256) o[c] = (_Float16)(r[c] * inv);
}

// ---------------------------------------------------------------------------
// WMMA GEMM-NT: C[M,N] = A[M,K] * B[N,K]^T  (f16 in, f32 out)
// One wave (32 threads) per block; each wave does a 16(M) x 64(N) strip,
// reusing one A fragment across 4 B fragments per 32-wide K step.
// blockIdx.z = batch, with element strides sA/sB/sC.
// ---------------------------------------------------------------------------
__global__ __launch_bounds__(32) void wmma_gemm_nt_kernel(
        const _Float16* __restrict__ A, const _Float16* __restrict__ Bm,
        float* __restrict__ C, int M, int N, int K,
        long long sA, long long sB, long long sC) {
    const int lane = threadIdx.x & 31;
    const int half = lane >> 4;      // 0: lanes 0-15, 1: lanes 16-31
    const int l16  = lane & 15;
    const long long bz = blockIdx.z;
    A  += bz * sA;  Bm += bz * sB;  C += bz * sC;

    const int n0 = blockIdx.x * 64;
    const int m0 = blockIdx.y * 16;
    const _Float16* aRow = A + (long long)(m0 + l16) * K;

    v8f acc[4] = {v8f{}, v8f{}, v8f{}, v8f{}};

    for (int k0 = 0; k0 < K; k0 += 32) {
        // A fragment: 16x32 fp16.  lanes 0-15: K {0..15}; lanes 16-31: K {16..31}
        v16h afrag;
        {
            v8h lo = *(const v8h*)(aRow + k0 + half * 16);
            v8h hi = *(const v8h*)(aRow + k0 + half * 16 + 8);
#pragma unroll
            for (int e = 0; e < 8; ++e) { afrag[e] = lo[e]; afrag[e + 8] = hi[e]; }
        }
        // 4 B fragments: 32x16 fp16 each. lane -> column n0+sub*16+l16; K contiguous.
#pragma unroll
        for (int sub = 0; sub < 4; ++sub) {
            const _Float16* bRow = Bm + (long long)(n0 + sub * 16 + l16) * K + k0 + half * 16;
            v8h lo = *(const v8h*)(bRow);
            v8h hi = *(const v8h*)(bRow + 8);
            v16h bfrag;
#pragma unroll
            for (int e = 0; e < 8; ++e) { bfrag[e] = lo[e]; bfrag[e + 8] = hi[e]; }
            acc[sub] = __builtin_amdgcn_wmma_f32_16x16x32_f16(
                false, afrag, false, bfrag, (short)0, acc[sub], false, false);
        }
    }

    // C layout: VGPR r, lanes 0-15 -> (M=r, N=lane); lanes 16-31 -> (M=8+r, N=lane-16)
    float* cBase = C + (long long)(m0 + half * 8) * N + n0 + l16;
#pragma unroll
    for (int sub = 0; sub < 4; ++sub)
#pragma unroll
        for (int r = 0; r < 8; ++r)
            cBase[(long long)r * N + sub * 16] = acc[sub][r];
}

// ---------------------------------------------------------------------------
// Per-row attention projections: AS[row]=h.att_src, AD[row]=h.att_dst
// ---------------------------------------------------------------------------
__global__ __launch_bounds__(256) void att_proj_kernel(const float* __restrict__ H,
                                                       const float* __restrict__ asv,
                                                       const float* __restrict__ adv,
                                                       float* __restrict__ AS,
                                                       float* __restrict__ AD) {
    __shared__ float r1[256], r2[256];
    const long long row = blockIdx.x;
    const float* h = H + row * (long long)DD;
    float s1 = 0.f, s2 = 0.f;
    for (int c = threadIdx.x; c < DD; c += 256) {
        float x = h[c]; s1 += x * asv[c]; s2 += x * adv[c];
    }
    r1[threadIdx.x] = s1; r2[threadIdx.x] = s2; __syncthreads();
    for (int w = 128; w > 0; w >>= 1) {
        if (threadIdx.x < w) { r1[threadIdx.x] += r1[threadIdx.x + w];
                               r2[threadIdx.x] += r2[threadIdx.x + w]; }
        __syncthreads();
    }
    if (threadIdx.x == 0) { AS[row] = r1[0]; AD[row] = r2[0]; }
}

__device__ __forceinline__ float lrelu(float x) { return x >= 0.f ? x : NEG_SLOPE * x; }

// ---------------------------------------------------------------------------
// GAT aggregate for the reading-order chain graph. Incoming edges of node j:
// (j-1)->j [if j>0], j->j twice [once if j==0], (j+1)->j [if j<n-1].
// One block per row; 3-point stencil with softmax coefficients.
// ---------------------------------------------------------------------------
__global__ __launch_bounds__(256) void gat_combine_kernel(const float* __restrict__ H,
                                                          const float* __restrict__ AS,
                                                          const float* __restrict__ AD,
                                                          const float* __restrict__ bias,
                                                          float* __restrict__ G) {
    const long long row = blockIdx.x;
    const int jj = (int)(row % NN);
    const bool hasPrev = (jj >= 1);
    const bool hasNext = (jj <= NN - 2);

    const float ad_j = AD[row];
    const float e_self = lrelu(AS[row] + ad_j);
    const float e_prev = hasPrev ? lrelu(AS[row - 1] + ad_j) : -3.0e38f;
    const float e_next = hasNext ? lrelu(AS[row + 1] + ad_j) : -3.0e38f;
    const float selfCnt = hasPrev ? 2.0f : 1.0f;   // duplicate self edge for j>=1

    const float emax = fmaxf(e_self, fmaxf(e_prev, e_next));
    const float wp = hasPrev ? expf(e_prev - emax) : 0.f;
    const float wn = hasNext ? expf(e_next - emax) : 0.f;
    const float wsf = expf(e_self - emax);
    const float denom = wp + wn + selfCnt * wsf;
    const float a_p = wp / denom, a_n = wn / denom, a_s = selfCnt * wsf / denom;

    const float* hj = H + row * (long long)DD;
    float* g = G + row * (long long)DD;
    for (int c = threadIdx.x; c < DD; c += 256) {
        float o = a_s * hj[c];
        if (hasPrev) o += a_p * hj[c - DD];
        if (hasNext) o += a_n * hj[c + DD];
        g[c] = o + bias[c];
    }
}

// ---------------------------------------------------------------------------
// Jonker-Volgenant LAP, one block (512 threads) per 384x384 cost matrix.
// cost(i,j) = 1 - |dot(i,j)| from the normalized-GEMM output.
// Writes out[mat] = mean_i cost(i, col4row[i]).
// ---------------------------------------------------------------------------
__global__ __launch_bounds__(512) void lap_kernel(const float* __restrict__ Call,
                                                  float* __restrict__ out) {
    __shared__ float u[NN], v[NN], shortest[NN];
    __shared__ int   path[NN], row4col[NN], col4row[NN];
    __shared__ unsigned char SR[NN], SC[NN];
    __shared__ float rv[512];
    __shared__ int   ri[512];
    __shared__ float s_minval;
    __shared__ int   s_i, s_sink;

    const int tid = threadIdx.x;
    const float* Cm = Call + (long long)blockIdx.x * NN * NN;
    const float INF = __builtin_inff();

    if (tid < NN) { u[tid] = 0.f; v[tid] = 0.f; row4col[tid] = -1; col4row[tid] = -1; }
    __syncthreads();

    for (int cur = 0; cur < NN; ++cur) {
        if (tid < NN) { shortest[tid] = INF; SC[tid] = 0; SR[tid] = 0; }
        if (tid == 0) { s_i = cur; s_sink = -1; s_minval = 0.f; }
        __syncthreads();

        while (true) {
            const int i = s_i;
            const float minval = s_minval;
            if (tid == 0) SR[i] = 1;
            if (tid < NN && !SC[tid]) {
                float cst = 1.0f - fabsf(Cm[(long long)i * NN + tid]);
                float r = minval + cst - u[i] - v[tid];
                if (r < shortest[tid]) { shortest[tid] = r; path[tid] = i; }
            }
            __syncthreads();

            rv[tid] = (tid < NN && !SC[tid]) ? shortest[tid] : INF;
            ri[tid] = tid;
            __syncthreads();
            for (int w = 256; w > 0; w >>= 1) {
                if (tid < w && rv[tid + w] < rv[tid]) { rv[tid] = rv[tid + w]; ri[tid] = ri[tid + w]; }
                __syncthreads();
            }
            if (tid == 0) {
                const int k = ri[0];
                s_minval = rv[0];
                SC[k] = 1;
                if (row4col[k] == -1) s_sink = k; else s_i = row4col[k];
            }
            __syncthreads();
            if (s_sink != -1) break;
        }

        const float minval = s_minval;
        const int   sink   = s_sink;
        __syncthreads();

        // dual updates
        if (tid == 0) u[cur] += minval;
        if (tid < NN && tid != cur && SR[tid]) u[tid] += minval - shortest[col4row[tid]];
        if (tid < NN && SC[tid])               v[tid] -= minval - shortest[tid];
        __syncthreads();

        // augment along the alternating path
        if (tid == 0) {
            int j = sink;
            while (true) {
                const int i = path[j];
                row4col[j] = i;
                const int t = col4row[i];
                col4row[i] = j;
                j = t;
                if (i == cur) break;
            }
        }
        __syncthreads();
    }

    // mean of picked entries
    rv[tid] = (tid < NN) ? (1.0f - fabsf(Cm[(long long)tid * NN + col4row[tid]])) : 0.f;
    __syncthreads();
    for (int w = 256; w > 0; w >>= 1) {
        if (tid < w) rv[tid] += rv[tid + w];
        __syncthreads();
    }
    if (tid == 0) out[blockIdx.x] = rv[0] / (float)NN;
}

// ---------------------------------------------------------------------------
// Host launcher
// ---------------------------------------------------------------------------
extern "C" void kernel_launch(void* const* d_in, const int* in_sizes, int n_in,
                              void* d_out, int out_size, void* d_ws, size_t ws_size,
                              hipStream_t stream) {
    const float* t1   = (const float*)d_in[0];   // (8,384,1024)
    const float* t2   = (const float*)d_in[1];   // (8,384,1024)
    const float* Wm   = (const float*)d_in[2];   // (1024,1024)
    const float* asv  = (const float*)d_in[3];   // (1024)
    const float* adv  = (const float*)d_in[4];   // (1024)
    const float* bias = (const float*)d_in[5];   // (1024)
    float* out = (float*)d_out;                  // local(8) || ctx(8)

    char* ws = (char*)d_ws;
    const long long ROWF = (long long)NROW * DD;     // 6,291,456 elements
    const long long HALF = (long long)NB * NN * DD;  // 3,145,728 elements (one tensor)

    // workspace layout (bytes), all 256-aligned; buffers reused stream-sequentially
    const size_t o_X16 = 0;                               // f16 inputs -> later normalized-input f16
    const size_t o_W16 = o_X16 + (size_t)ROWF * 2;        // 12,582,912
    const size_t o_H   = o_W16 + (size_t)DD * DD * 2;     // +2,097,152
    const size_t o_AS  = o_H   + (size_t)ROWF * 4;        // +25,165,824
    const size_t o_AD  = o_AS  + (size_t)NROW * 4;
    const size_t o_G   = o_AD  + (size_t)NROW * 4;        // fp32 GAT out -> later COST matrices

    _Float16* X16 = (_Float16*)(ws + o_X16);
    _Float16* W16 = (_Float16*)(ws + o_W16);
    float*    Hf  = (float*)   (ws + o_H);
    float*    ASp = (float*)   (ws + o_AS);
    float*    ADp = (float*)   (ws + o_AD);
    float*    Gf  = (float*)   (ws + o_G);
    _Float16* N16 = (_Float16*)(ws + o_H);   // normalized GAT f16 (reuses H region)
    float*    COST= (float*)   (ws + o_G);   // 16 cost matrices (reuses G region)

    // 1) fp16 copies of inputs and W
    cvt_f16_kernel<<<(unsigned)((HALF + 255) / 256), 256, 0, stream>>>(t1, X16, HALF);
    cvt_f16_kernel<<<(unsigned)((HALF + 255) / 256), 256, 0, stream>>>(t2, X16 + HALF, HALF);
    cvt_f16_kernel<<<(unsigned)(((long long)DD * DD + 255) / 256), 256, 0, stream>>>(Wm, W16, (long long)DD * DD);

    // 2) H = X @ W^T  (M=6144, N=K=1024)
    {
        dim3 grid(DD / 64, NROW / 16, 1);
        wmma_gemm_nt_kernel<<<grid, 32, 0, stream>>>(X16, W16, Hf, NROW, DD, DD, 0, 0, 0);
    }

    // 3) attention projections
    att_proj_kernel<<<NROW, 256, 0, stream>>>(Hf, asv, adv, ASp, ADp);

    // 4) GAT stencil aggregate
    gat_combine_kernel<<<NROW, 256, 0, stream>>>(Hf, ASp, ADp, bias, Gf);

    // 5) row-normalized f16 operands (X16 region now free; H region free after 4)
    rownorm_f16_kernel<<<NB * NN, 256, 0, stream>>>(t1, X16);             // An1
    rownorm_f16_kernel<<<NB * NN, 256, 0, stream>>>(t2, X16 + HALF);      // Bn1
    rownorm_f16_kernel<<<NB * NN, 256, 0, stream>>>(Gf, N16);             // An2 (g1)
    rownorm_f16_kernel<<<NB * NN, 256, 0, stream>>>(Gf + HALF, N16 + HALF); // Bn2 (g2)

    // 6) batched cosine-similarity GEMMs -> raw dot products (16 x 384x384)
    {
        dim3 grid(NN / 64, NN / 16, NB);
        const long long sAB = (long long)NN * DD, sC = (long long)NN * NN;
        wmma_gemm_nt_kernel<<<grid, 32, 0, stream>>>(X16, X16 + HALF, COST,
                                                     NN, NN, DD, sAB, sAB, sC);
        wmma_gemm_nt_kernel<<<grid, 32, 0, stream>>>(N16, N16 + HALF, COST + (long long)NB * NN * NN,
                                                     NN, NN, DD, sAB, sAB, sC);
    }

    // 7) Hungarian assignment + mean of picked costs: out[0..7]=local, out[8..15]=ctx
    lap_kernel<<<16, 512, 0, stream>>>(COST, out);

    (void)in_sizes; (void)n_in; (void)out_size; (void)ws_size;
}
